// SpatialGraphEncoder_14534169329721
// MI455X (gfx1250) — matrix operand order repaired and checked by reading, not compiled, not run on hardware
//
#include <hip/hip_runtime.h>
#include <hip/hip_bf16.h>
#include <math.h>
#include <stdint.h>

// ---------------------------------------------------------------------------
// SpatialGraphEncoder for MI455X (gfx1250, wave32, WMMA).
// Dense projections: bf16 WMMA GEMM (v_wmma_f32_16x16x32_bf16, fp32 accum),
// double-buffered LDS ping-pong, weights pre-transposed to bf16 once per
// launch, fragments loaded with 16B ds_load_b128 in native WMMA layout.
// ---------------------------------------------------------------------------

typedef __attribute__((ext_vector_type(16))) __bf16 v16bf;
typedef __attribute__((ext_vector_type(8)))  __bf16 v8bf;
typedef __attribute__((ext_vector_type(4)))  __bf16 v4bf;
typedef __attribute__((ext_vector_type(8)))  float  v8f;

constexpr int NN   = 16384;   // nodes
constexpr int DD   = 256;     // model dim
constexpr int BB   = 8;       // batches
constexpr int MM   = 2048;    // nodes per batch
constexpr int HH   = 8;       // heads
constexpr int EE   = 262144;  // edges
constexpr int NNZC = 524288;  // sparse obs
constexpr int KMC  = 16384;   // masked positions
constexpr int GC   = 8;       // global tokens

__device__ inline __bf16 f2bf(float f) { return (__bf16)f; }   // native cvt (RNE)

__device__ inline void atomicMaxF(float* addr, float val) {
  unsigned* ua = (unsigned*)addr;
  unsigned old = *ua;
  while (true) {
    float f = __uint_as_float(old);
    if (f >= val) break;
    unsigned assumed = old;
    old = atomicCAS(ua, assumed, __float_as_uint(val));
    if (old == assumed) break;
  }
}

// ---------------------------------------------------------------------------
// WMMA GEMM: C[M,N] = act(A[M,K] @ Wt^T + bias)
//   A  : fp32 row-major [M,K]   (K multiple of 32, rows 16B aligned)
//   Wt : bf16 row-major [N,K]   (pre-transposed weight)
// block = 256 threads (8 waves), block tile 128x64, wave tile 32x32,
// K step 32, double-buffered LDS (one barrier per K-step).
// ---------------------------------------------------------------------------
constexpr int BMT = 128, BNT = 64, BKT = 32;
constexpr int LDA = 40;   // bf16 elems: 80B row stride -> 16B aligned, bank-safe
constexpr int LDB = 40;

union Frag { v16bf v; v8bf h[2]; };

template <bool GUARD_M>
__global__ __launch_bounds__(256)
void k_gemm_bf16(const float* __restrict__ A, const __bf16* __restrict__ Wt,
                 const float* __restrict__ bias, float* __restrict__ C,
                 int M, int K, int N, int act)
{
  __shared__ __bf16 As[2][BMT * LDA];   // 2 x 10.0 KB
  __shared__ __bf16 Bs[2][BNT * LDB];   // 2 x  5.0 KB
  const int tid  = threadIdx.x;
  const int lane = tid & 31;
  const int wave = tid >> 5;
  const int wr   = wave >> 1;           // 0..3 : 32-row band
  const int wc   = wave & 1;            // 0..1 : 32-col band
  const int m0   = blockIdx.x * BMT;
  const int n0   = blockIdx.y * BNT;

  v8f acc[2][2] = {{{}, {}}, {{}, {}}};

  const int ml = lane & 15;
  const int ka = (lane < 16) ? 0 : 8;   // A frag K base (ISA 16-bit A 16x32)
  const int kb = (lane < 16) ? 0 : 8;   // B frag K base (same k order as A)
  const int br = tid >> 2;              // B stage row 0..63
  const int bc = (tid & 3) * 8;         // B stage col 0,8,16,24

  float4 fA[4];
  v8bf   fB;

  auto loadTile = [&](int k0) {
    #pragma unroll
    for (int i = 0; i < 4; ++i) {
      int chunk = i * 256 + tid;
      int r = chunk >> 3;               // 0..127
      int c = (chunk & 7) * 4;          // 0,4,...,28
      if (!GUARD_M || (m0 + r) < M)
        fA[i] = *(const float4*)(A + (size_t)(m0 + r) * K + k0 + c);
      else
        fA[i] = make_float4(0.f, 0.f, 0.f, 0.f);
    }
    fB = *(const v8bf*)(Wt + (size_t)(n0 + br) * K + k0 + bc);
  };
  auto storeTile = [&](int buf) {
    #pragma unroll
    for (int i = 0; i < 4; ++i) {
      int chunk = i * 256 + tid;
      int r = chunk >> 3;
      int c = (chunk & 7) * 4;
      v4bf p;
      p[0] = f2bf(fA[i].x); p[1] = f2bf(fA[i].y);
      p[2] = f2bf(fA[i].z); p[3] = f2bf(fA[i].w);
      *(v4bf*)&As[buf][r * LDA + c] = p;
    }
    *(v8bf*)&Bs[buf][br * LDB + bc] = fB;
  };

  const int nk = K / BKT;
  loadTile(0);
  storeTile(0);
  __syncthreads();

  int buf = 0;
  for (int kt = 0; kt < nk; ++kt) {
    if (kt + 1 < nk) loadTile((kt + 1) * BKT);   // global loads in flight
    if (kt + 2 < nk)                              // prefetch tile after next
      __builtin_prefetch(A + (size_t)(m0 + (tid >> 3)) * K + (kt + 2) * BKT, 0, 1);

    // fragments: two 16B LDS vector loads each
    Frag a[2], b[2];
    #pragma unroll
    for (int s = 0; s < 2; ++s) {
      const __bf16* ap = &As[buf][(wr * 32 + s * 16 + ml) * LDA + ka];
      a[s].h[0] = *(const v8bf*)(ap);        // elems 0..7  : k = ka+0..7
      a[s].h[1] = *(const v8bf*)(ap + 16);   // elems 8..15 : k = ka+16..23
      const __bf16* bp = &Bs[buf][(wc * 32 + s * 16 + ml) * LDB + kb];
      b[s].h[0] = *(const v8bf*)(bp);        // k = kb+0..7
      b[s].h[1] = *(const v8bf*)(bp + 16);   // k = kb+16..23
    }
    #pragma unroll
    for (int s = 0; s < 2; ++s)
      #pragma unroll
      for (int t = 0; t < 2; ++t)
        acc[s][t] = __builtin_amdgcn_wmma_f32_16x16x32_bf16(
            false, a[s].v, false, b[t].v, (short)0, acc[s][t], false, false);

    if (kt + 1 < nk) storeTile(buf ^ 1);          // fill other buffer
    buf ^= 1;
    __syncthreads();
  }

  // epilogue: element r -> row r + 8*(lane>=16), col lane&15
  const int nlo = lane & 15;
  const int mhi = (lane >= 16) ? 8 : 0;
  #pragma unroll
  for (int s = 0; s < 2; ++s) {
    #pragma unroll
    for (int r = 0; r < 8; ++r) {
      int gm = m0 + wr * 32 + s * 16 + mhi + r;
      if (GUARD_M && gm >= M) continue;
      #pragma unroll
      for (int t = 0; t < 2; ++t) {
        int gn = n0 + wc * 32 + t * 16 + nlo;
        float v = acc[s][t][r];
        if (bias) v += bias[gn];
        if (act == 1)      v = fmaxf(v, 0.f);                               // relu
        else if (act == 2) v = 0.5f * v * (1.f + erff(v * 0.70710678118f)); // gelu
        C[(size_t)gm * N + gn] = v;
      }
    }
  }
}

// weight pre-transpose: Wt[n*Kp + k] = bf16(W[k*ldw + n]), zero-pad k >= K
__global__ void k_transpose_w(const float* __restrict__ W, __bf16* __restrict__ Wt,
                              int K, int N, int ldw, int Kp)
{
  int i = blockIdx.x * 256 + threadIdx.x;
  if (i >= N * Kp) return;
  int n = i / Kp, k = i % Kp;
  float v = (k < K) ? W[(size_t)k * ldw + n] : 0.f;
  Wt[(size_t)n * Kp + k] = f2bf(v);
}

// zero-pad spatial_enc (N x 40) -> (N x 64)
__global__ void k_pad_spatial(const float* __restrict__ s, float* __restrict__ sp) {
  size_t i = (size_t)blockIdx.x * 256 + threadIdx.x;   // over N*64
  size_t r = i >> 6; int c = (int)(i & 63);
  sp[i] = (c < 40) ? s[r * 40 + c] : 0.f;
}

// ---------------------------------------------------------------------------
// elementwise / scatter / reduction kernels
// ---------------------------------------------------------------------------
__global__ void k_fill(float* p, size_t n, float v) {
  size_t i = (size_t)blockIdx.x * 256 + threadIdx.x;
  if (i < n) p[i] = v;
}

__global__ void k_scatter_obs(const int* __restrict__ rows, const int* __restrict__ cols,
                              const float* __restrict__ vals, const float* __restrict__ GW,
                              float* __restrict__ gene)
{
  int e = blockIdx.x, d = threadIdx.x;
  int r = rows[e], c = cols[e];
  atomicAdd(&gene[(size_t)r * DD + d], vals[e] * GW[(size_t)c * DD + d]);
}

__global__ void k_scatter_masked(const int* __restrict__ mpos, const int* __restrict__ rows,
                                 const int* __restrict__ cols, const float* __restrict__ vals,
                                 const float* __restrict__ GW, float* __restrict__ gene,
                                 float* __restrict__ mcount)
{
  int i = blockIdx.x, d = threadIdx.x;
  int p = mpos[i];
  int r = rows[p], c = cols[p];
  atomicAdd(&gene[(size_t)r * DD + d], -vals[p] * GW[(size_t)c * DD + d]);
  if (d == 0) atomicAdd(&mcount[r], 1.0f);
}

__global__ void k_gelu_bias(float* g, const float* b) {
  size_t i = (size_t)blockIdx.x * 256 + threadIdx.x;
  float v = g[i] + b[i & (DD - 1)];
  g[i] = 0.5f * v * (1.f + erff(v * 0.70710678118f));
}

__global__ void k_concat2(const float* a, const float* b, float* c) {
  size_t i = (size_t)blockIdx.x * 256 + threadIdx.x;   // over N*512
  size_t r = i >> 9; int col = (int)(i & 511);
  c[i] = (col < DD) ? a[r * DD + col] : b[r * DD + (col - DD)];
}

__global__ void k_maskfeat(const float* mc, const float* me, float* mf) {
  size_t i = (size_t)blockIdx.x * 256 + threadIdx.x;   // over N*D
  mf[i] = mc[i >> 8] * me[i & (DD - 1)];
}

__global__ void k_fuse_gate(const float* gp, const float* mg, const float* gene,
                            const float* spat, float* xn) {
  size_t i = (size_t)blockIdx.x * 256 + threadIdx.x;   // over N*D
  float g = 1.f / (1.f + expf(-(gp[i] + mg[i])));
  xn[i] = g * gene[i] + (1.f - g) * spat[i];
}

__global__ void k_init_gh(const float* gtg, const float* gtl, float* gh) {
  int i = blockIdx.x * 256 + threadIdx.x;              // B*G*D = 16384
  int d = i & (DD - 1), g = (i >> 8) & 7;
  gh[i] = (g < 2) ? gtg[g * DD + d] : gtl[(g - 2) * DD + d];
}

// ---- graph (edge) attention ----
__global__ void k_edge_score(const int* __restrict__ ei, const float* __restrict__ q,
                             const float* __restrict__ k, float* __restrict__ sc,
                             float* __restrict__ mbuf)
{
  int i = blockIdx.x * 256 + threadIdx.x;              // E*H
  int e = i >> 3, h = i & 7;
  int s = ei[e], d = ei[EE + e];
  const float* qp = q + (size_t)d * DD + h * 32;
  const float* kp = k + (size_t)s * DD + h * 32;
  float acc = 0.f;
  #pragma unroll
  for (int j = 0; j < 32; ++j) acc += qp[j] * kp[j];
  acc *= 0.17677669529663687f;                         // 1/sqrt(32)
  sc[i] = acc;
  atomicMaxF(&mbuf[(size_t)d * HH + h], acc);
}

__global__ void k_edge_exp(const int* __restrict__ ei, float* __restrict__ sc,
                           const float* __restrict__ mbuf, float* __restrict__ zbuf)
{
  int i = blockIdx.x * 256 + threadIdx.x;              // E*H
  int e = i >> 3, h = i & 7;
  int d = ei[EE + e];
  float v = expf(sc[i] - mbuf[(size_t)d * HH + h]);
  sc[i] = v;
  atomicAdd(&zbuf[(size_t)d * HH + h], v);
}

__global__ void k_edge_agg(const int* __restrict__ ei, const float* __restrict__ sc,
                           const float* __restrict__ zbuf, const float* __restrict__ v,
                           float* __restrict__ agg)
{
  int e = blockIdx.x, t = threadIdx.x;
  int h = t >> 5, dd = t & 31;
  int s = ei[e], d = ei[EE + e];
  float alpha = sc[(size_t)e * HH + h] / zbuf[(size_t)d * HH + h];
  atomicAdd(&agg[(size_t)d * DD + h * 32 + dd], alpha * v[(size_t)s * DD + h * 32 + dd]);
}

// ---- beta-gate + residual + LayerNorm (one block per row, 256 threads) ----
__global__ __launch_bounds__(256)
void k_beta_ln1(const float* __restrict__ xn, const float* __restrict__ skip,
                const float* __restrict__ agg, const float* __restrict__ wbeta,
                const float* __restrict__ g, const float* __restrict__ bb,
                float* __restrict__ out)
{
  int r = blockIdx.x, t = threadIdx.x;
  __shared__ float red[256];
  float s = skip[(size_t)r * DD + t];
  float a = agg[(size_t)r * DD + t];
  float p = s * wbeta[t] + a * wbeta[DD + t] + (s - a) * wbeta[2 * DD + t];
  red[t] = p; __syncthreads();
  for (int st = 128; st > 0; st >>= 1) { if (t < st) red[t] += red[t + st]; __syncthreads(); }
  float beta = 1.f / (1.f + expf(-red[0])); __syncthreads();
  float y = xn[(size_t)r * DD + t] + beta * s + (1.f - beta) * a;
  red[t] = y; __syncthreads();
  for (int st = 128; st > 0; st >>= 1) { if (t < st) red[t] += red[t + st]; __syncthreads(); }
  float mu = red[0] * (1.f / DD); __syncthreads();
  float c = y - mu;
  red[t] = c * c; __syncthreads();
  for (int st = 128; st > 0; st >>= 1) { if (t < st) red[t] += red[t + st]; __syncthreads(); }
  float var = red[0] * (1.f / DD);
  out[(size_t)r * DD + t] = c * rsqrtf(var + 1e-5f) * g[t] + bb[t];
}

__global__ __launch_bounds__(256)
void k_add_ln(const float* __restrict__ x, const float* __restrict__ dx,
              const float* __restrict__ g, const float* __restrict__ bb,
              float* __restrict__ out)
{
  int r = blockIdx.x, t = threadIdx.x;
  __shared__ float red[256];
  float y = x[(size_t)r * DD + t] + dx[(size_t)r * DD + t];
  red[t] = y; __syncthreads();
  for (int st = 128; st > 0; st >>= 1) { if (t < st) red[t] += red[t + st]; __syncthreads(); }
  float mu = red[0] * (1.f / DD); __syncthreads();
  float c = y - mu;
  red[t] = c * c; __syncthreads();
  for (int st = 128; st > 0; st >>= 1) { if (t < st) red[t] += red[t + st]; __syncthreads(); }
  float var = red[0] * (1.f / DD);
  out[(size_t)r * DD + t] = c * rsqrtf(var + 1e-5f) * g[t] + bb[t];
}

// ---- global tokens attend to nodes (softmax over 2048 keys) ----
__global__ __launch_bounds__(256)
void k_mha_g2n(const float* __restrict__ qh, const float* __restrict__ kh,
               const float* __restrict__ vh, float* __restrict__ obuf,
               float* __restrict__ attn)
{
  int bi = blockIdx.x;                 // b*64 + q*8 + h  (512 blocks)
  int h = bi & 7, q = (bi >> 3) & 7, b = bi >> 6;
  __shared__ float qv[32];
  __shared__ float red[256];
  __shared__ float oacc[32];
  int tid = threadIdx.x;
  if (tid < 32) { qv[tid] = qh[((size_t)(b * GC + q)) * DD + h * 32 + tid]; oacc[tid] = 0.f; }
  __syncthreads();
  float dots[8];
  float mx = -1e30f;
  #pragma unroll
  for (int it = 0; it < 8; ++it) {
    int kk = it * 256 + tid;
    const float* kp = kh + ((size_t)(b * MM + kk)) * DD + h * 32;
    float d = 0.f;
    #pragma unroll
    for (int j = 0; j < 32; ++j) d += qv[j] * kp[j];
    d *= 0.17677669529663687f;
    dots[it] = d;
    mx = fmaxf(mx, d);
  }
  red[tid] = mx; __syncthreads();
  for (int st = 128; st > 0; st >>= 1) { if (tid < st) red[tid] = fmaxf(red[tid], red[tid + st]); __syncthreads(); }
  mx = red[0]; __syncthreads();
  float lsum = 0.f;
  #pragma unroll
  for (int it = 0; it < 8; ++it) { dots[it] = expf(dots[it] - mx); lsum += dots[it]; }
  red[tid] = lsum; __syncthreads();
  for (int st = 128; st > 0; st >>= 1) { if (tid < st) red[tid] += red[tid + st]; __syncthreads(); }
  float inv = 1.f / red[0];
  #pragma unroll
  for (int it = 0; it < 8; ++it) {
    int kk = it * 256 + tid;
    float p = dots[it] * inv;
    atomicAdd(&attn[((size_t)b * GC + q) * MM + kk], p * 0.125f);  // mean over H
    const float* vp = vh + ((size_t)(b * MM + kk)) * DD + h * 32;
    #pragma unroll
    for (int j = 0; j < 32; ++j) atomicAdd(&oacc[j], p * vp[j]);
  }
  __syncthreads();
  if (tid < 32) obuf[((size_t)(b * GC + q)) * DD + h * 32 + tid] = oacc[tid];
}

// ---- nodes attend to 8 global tokens (thread per (node, head)) ----
__global__ __launch_bounds__(256)
void k_mha_n2g(const float* __restrict__ qh, const float* __restrict__ kh,
               const float* __restrict__ vh, float* __restrict__ o)
{
  int i = blockIdx.x * 256 + threadIdx.x;       // N*H
  int n = i >> 3, h = i & 7;
  int b = n >> 11;                              // n / 2048
  const float* qp = qh + (size_t)n * DD + h * 32;
  float sc[8]; float mx = -1e30f;
  #pragma unroll
  for (int t = 0; t < 8; ++t) {
    const float* kp = kh + (size_t)(b * GC + t) * DD + h * 32;
    float d = 0.f;
    #pragma unroll
    for (int j = 0; j < 32; ++j) d += qp[j] * kp[j];
    d *= 0.17677669529663687f;
    sc[t] = d; mx = fmaxf(mx, d);
  }
  float z = 0.f;
  #pragma unroll
  for (int t = 0; t < 8; ++t) { sc[t] = expf(sc[t] - mx); z += sc[t]; }
  float inv = 1.f / z;
  float* op = o + (size_t)n * DD + h * 32;
  #pragma unroll
  for (int j = 0; j < 32; ++j) {
    float acc = 0.f;
    #pragma unroll
    for (int t = 0; t < 8; ++t) acc += sc[t] * vh[(size_t)(b * GC + t) * DD + h * 32 + j];
    op[j] = acc * inv;
  }
}

// ---- token diversity regularizer (rows 2..7 of attn per batch) ----
__global__ __launch_bounds__(256)
void k_token_reg(const float* __restrict__ attn, float* __restrict__ reg)
{
  int b = blockIdx.x, tid = threadIdx.x;
  __shared__ float red[256];
  __shared__ float srow[6];
  const float* Pb = attn + (size_t)b * GC * MM;
  for (int i = 0; i < 6; ++i) {
    const float* row = Pb + (size_t)(2 + i) * MM;
    float p = 0.f;
    for (int kk = tid; kk < MM; kk += 256) p += row[kk];
    red[tid] = p; __syncthreads();
    for (int st = 128; st > 0; st >>= 1) { if (tid < st) red[tid] += red[tid + st]; __syncthreads(); }
    if (tid == 0) srow[i] = red[0] + 1e-8f;
    __syncthreads();
  }
  float offsum = 0.f;
  for (int i = 0; i < 6; ++i)
    for (int j = 0; j < 6; ++j) {
      if (i == j) continue;
      const float* ri = Pb + (size_t)(2 + i) * MM;
      const float* rj = Pb + (size_t)(2 + j) * MM;
      float p = 0.f;
      for (int kk = tid; kk < MM; kk += 256) p += ri[kk] * rj[kk];
      red[tid] = p; __syncthreads();
      for (int st = 128; st > 0; st >>= 1) { if (tid < st) red[tid] += red[tid + st]; __syncthreads(); }
      if (tid == 0) offsum += red[0] / (srow[i] * srow[j]);
      __syncthreads();
    }
  if (tid == 0) atomicAdd(reg, offsum * (1.f / 30.f) * (1.f / (BB * 2.f)));
}

__global__ void k_write_out(const float* xn, const float* gh, const float* reg,
                            float* out, size_t total)
{
  size_t i = (size_t)blockIdx.x * 256 + threadIdx.x;
  if (i >= total) return;
  const size_t ND = (size_t)NN * DD, GHN = (size_t)BB * GC * DD;
  if (i < ND)            out[i] = xn[i];
  else if (i < ND + GHN) out[i] = gh[i - ND];
  else                   out[i] = reg[0];
}

// ---------------------------------------------------------------------------
// host orchestration
// ---------------------------------------------------------------------------
struct LayerP {
  const float *a1_in_b, *a1_in_w, *a1_out_b, *a1_out_w;
  const float *a2_in_b, *a2_in_w, *a2_out_b, *a2_out_w;
  const float *b1, *b2, *bk, *bq, *bskip, *bv;
  const float *n1b, *n1g, *n2b, *n2g, *ngb, *ngg, *nngb, *nngg;
  const float *w1, *w2, *wbeta, *wk, *wq, *wskip, *wv;
};

// params flattened as a jax pytree: dict keys sorted, lists in order.
// global: 9 fus_b 10 fus_w 11 gene_bias 12 gene_weight 13 gt_g 14 gt_l
//         [15..43] layer0, [44..72] layer1, 73 mask_emb 74 mg_b 75 mg_w 76 sp_b 77 sp_w
static LayerP get_layer(void* const* d_in, int base) {
  auto f = [&](int i) { return (const float*)d_in[base + i]; };
  LayerP p;
  p.a1_in_b = f(0);  p.a1_in_w = f(1);  p.a1_out_b = f(2);  p.a1_out_w = f(3);
  p.a2_in_b = f(4);  p.a2_in_w = f(5);  p.a2_out_b = f(6);  p.a2_out_w = f(7);
  p.b1 = f(8);  p.b2 = f(9);  p.bk = f(10); p.bq = f(11); p.bskip = f(12); p.bv = f(13);
  p.n1b = f(14); p.n1g = f(15); p.n2b = f(16); p.n2g = f(17);
  p.ngb = f(18); p.ngg = f(19); p.nngb = f(20); p.nngg = f(21);
  p.w1 = f(22); p.w2 = f(23); p.wbeta = f(24); p.wk = f(25); p.wq = f(26);
  p.wskip = f(27); p.wv = f(28);
  return p;
}

static void gemm(hipStream_t st, const float* A, const __bf16* Wt, const float* bias,
                 float* C, int M, int K, int N, int act)
{
  dim3 grid((M + BMT - 1) / BMT, N / BNT);
  if (M % BMT == 0)
    k_gemm_bf16<false><<<grid, 256, 0, st>>>(A, Wt, bias, C, M, K, N, act);
  else
    k_gemm_bf16<true><<<grid, 256, 0, st>>>(A, Wt, bias, C, M, K, N, act);
}

static void fillf(hipStream_t st, float* p, size_t n, float v) {
  k_fill<<<(unsigned)((n + 255) / 256), 256, 0, st>>>(p, n, v);
}

static void xposew(hipStream_t st, const float* W, __bf16* Wt, int K, int N, int ldw, int Kp) {
  int tot = N * Kp;
  k_transpose_w<<<(tot + 255) / 256, 256, 0, st>>>(W, Wt, K, N, ldw, Kp);
}

extern "C" void kernel_launch(void* const* d_in, const int* in_sizes, int n_in,
                              void* d_out, int out_size, void* d_ws, size_t ws_size,
                              hipStream_t stream)
{
  (void)in_sizes; (void)n_in; (void)out_size; (void)ws_size;
  const int*   edge_index = (const int*)d_in[1];
  const float* spatial    = (const float*)d_in[2];
  const int*   obs_rows   = (const int*)d_in[4];
  const int*   obs_cols   = (const int*)d_in[5];
  const float* obs_vals   = (const float*)d_in[6];
  const int*   mpos       = (const int*)d_in[7];
  const float* fus_b    = (const float*)d_in[9];
  const float* fus_w    = (const float*)d_in[10];
  const float* gene_b   = (const float*)d_in[11];
  const float* gene_w   = (const float*)d_in[12];
  const float* gt_g     = (const float*)d_in[13];
  const float* gt_l     = (const float*)d_in[14];
  const float* mask_emb = (const float*)d_in[73];
  const float* mg_b     = (const float*)d_in[74];
  const float* mg_w     = (const float*)d_in[75];
  const float* sp_b     = (const float*)d_in[76];
  const float* sp_w     = (const float*)d_in[77];

  // -------- workspace carve: fp32 activations --------
  float* w = (float*)d_ws;
  auto alloc = [&](size_t n) { float* p = w; w += n; return p; };
  const size_t ND = (size_t)NN * DD;
  float* xn     = alloc(ND);
  float* gene   = alloc(ND);
  float* spat   = alloc(ND);
  float* spad   = alloc((size_t)NN * 64);   // K-padded spatial input
  float* cat2   = alloc(2 * ND);            // reused as ffn1
  float* maskf  = alloc(ND);                // reused as ffn2
  float* gatep  = alloc(ND);
  float* mgate  = alloc(ND);
  float* qb     = alloc(ND);                // reused as kh1
  float* kb     = alloc(ND);                // reused as vh1
  float* vb     = alloc(ND);                // reused as qh2
  float* skipb  = alloc(ND);                // reused as o2
  float* aggb   = alloc(ND);                // reused as n_out
  float* scores = alloc((size_t)EE * HH);
  float* mbuf   = alloc((size_t)NN * HH);
  float* zbuf   = alloc((size_t)NN * HH);
  float* mcount = alloc(NN);
  float* gh     = alloc((size_t)BB * GC * DD);
  float* ghtmp  = alloc((size_t)BB * GC * DD);
  float* qh1    = alloc((size_t)BB * GC * DD);
  float* obuf   = alloc((size_t)BB * GC * DD);
  float* kh2    = alloc((size_t)BB * GC * DD);
  float* vh2    = alloc((size_t)BB * GC * DD);
  float* attn   = alloc((size_t)BB * GC * MM);
  float* reg    = alloc(1);

  // -------- workspace carve: bf16 pre-transposed weights (16B aligned) --------
  uintptr_t up = (uintptr_t)w; up = (up + 15) & ~(uintptr_t)15;
  __bf16* wbp = (__bf16*)up;
  auto balloc = [&](size_t n) { __bf16* p = wbp; wbp += n; return p; };
  __bf16* sp_t  = balloc(256 * 64);
  __bf16* fus_t = balloc(256 * 512);
  __bf16* mg_t  = balloc(256 * 256);
  struct LW { __bf16 *wq, *wk, *wv, *wskip, *a1in, *a1out, *a2in, *a2out, *w1, *w2; } lw[2];
  for (int l = 0; l < 2; ++l) {
    lw[l].wq    = balloc(256 * 256);
    lw[l].wk    = balloc(256 * 256);
    lw[l].wv    = balloc(256 * 256);
    lw[l].wskip = balloc(256 * 256);
    lw[l].a1in  = balloc(768 * 256);
    lw[l].a1out = balloc(256 * 256);
    lw[l].a2in  = balloc(768 * 256);
    lw[l].a2out = balloc(256 * 256);
    lw[l].w1    = balloc(512 * 256);
    lw[l].w2    = balloc(256 * 512);
  }

  // -------- weight transposes (fp32 -> bf16, [K,N] -> [N,K]) --------
  xposew(stream, sp_w,  sp_t,  40,  256, 256, 64);
  xposew(stream, fus_w, fus_t, 512, 256, 256, 512);
  xposew(stream, mg_w,  mg_t,  256, 256, 256, 256);
  for (int l = 0; l < 2; ++l) {
    LayerP p = get_layer(d_in, 15 + 29 * l);
    xposew(stream, p.wq,       lw[l].wq,    256, 256, 256, 256);
    xposew(stream, p.wk,       lw[l].wk,    256, 256, 256, 256);
    xposew(stream, p.wv,       lw[l].wv,    256, 256, 256, 256);
    xposew(stream, p.wskip,    lw[l].wskip, 256, 256, 256, 256);
    xposew(stream, p.a1_in_w,  lw[l].a1in,  256, 768, 768, 256);
    xposew(stream, p.a1_out_w, lw[l].a1out, 256, 256, 256, 256);
    xposew(stream, p.a2_in_w,  lw[l].a2in,  256, 768, 768, 256);
    xposew(stream, p.a2_out_w, lw[l].a2out, 256, 256, 256, 256);
    xposew(stream, p.w1,       lw[l].w1,    256, 512, 512, 256);
    xposew(stream, p.w2,       lw[l].w2,    512, 256, 256, 512);
  }

  // -------- embedding --------
  fillf(stream, gene, ND, 0.f);
  fillf(stream, mcount, NN, 0.f);
  fillf(stream, reg, 1, 0.f);
  k_scatter_obs<<<NNZC, 256, 0, stream>>>(obs_rows, obs_cols, obs_vals, gene_w, gene);
  k_scatter_masked<<<KMC, 256, 0, stream>>>(mpos, obs_rows, obs_cols, obs_vals, gene_w,
                                            gene, mcount);
  k_gelu_bias<<<(unsigned)(ND / 256), 256, 0, stream>>>(gene, gene_b);
  k_pad_spatial<<<(unsigned)((size_t)NN * 64 / 256), 256, 0, stream>>>(spatial, spad);
  gemm(stream, spad, sp_t, sp_b, spat, NN, 64, DD, 0);
  k_concat2<<<(unsigned)(2 * ND / 256), 256, 0, stream>>>(gene, spat, cat2);
  gemm(stream, cat2, fus_t, fus_b, gatep, NN, 2 * DD, DD, 0);
  k_maskfeat<<<(unsigned)(ND / 256), 256, 0, stream>>>(mcount, mask_emb, maskf);
  gemm(stream, maskf, mg_t, mg_b, mgate, NN, DD, DD, 0);
  k_fuse_gate<<<(unsigned)(ND / 256), 256, 0, stream>>>(gatep, mgate, gene, spat, xn);
  k_init_gh<<<BB * GC * DD / 256, 256, 0, stream>>>(gt_g, gt_l, gh);

  // -------- layers --------
  for (int l = 0; l < 2; ++l) {
    LayerP p = get_layer(d_in, 15 + 29 * l);

    // graph attention projections (WMMA)
    gemm(stream, xn, lw[l].wq,    p.bq,    qb,    NN, DD, DD, 0);
    gemm(stream, xn, lw[l].wk,    p.bk,    kb,    NN, DD, DD, 0);
    gemm(stream, xn, lw[l].wv,    p.bv,    vb,    NN, DD, DD, 0);
    gemm(stream, xn, lw[l].wskip, p.bskip, skipb, NN, DD, DD, 0);
    fillf(stream, mbuf, (size_t)NN * HH, -1e30f);
    fillf(stream, zbuf, (size_t)NN * HH, 0.f);
    fillf(stream, aggb, ND, 0.f);
    k_edge_score<<<EE * HH / 256, 256, 0, stream>>>(edge_index, qb, kb, scores, mbuf);
    k_edge_exp<<<EE * HH / 256, 256, 0, stream>>>(edge_index, scores, mbuf, zbuf);
    k_edge_agg<<<EE, 256, 0, stream>>>(edge_index, scores, zbuf, vb, aggb);
    k_beta_ln1<<<NN, 256, 0, stream>>>(xn, skipb, aggb, p.wbeta, p.n1g, p.n1b, xn);

    // a1: global tokens attend to nodes  (in_w column slices = Wt row slices)
    fillf(stream, attn, (size_t)BB * GC * MM, 0.f);
    gemm(stream, gh, lw[l].a1in,                 p.a1_in_b,          qh1, BB * GC, DD, DD, 0);
    gemm(stream, xn, lw[l].a1in + 256 * 256,     p.a1_in_b + DD,     qb,  NN,      DD, DD, 0);
    gemm(stream, xn, lw[l].a1in + 2 * 256 * 256, p.a1_in_b + 2 * DD, kb,  NN,      DD, DD, 0);
    k_mha_g2n<<<BB * GC * HH, 256, 0, stream>>>(qh1, qb, kb, obuf, attn);
    gemm(stream, obuf, lw[l].a1out, p.a1_out_b, ghtmp, BB * GC, DD, DD, 0);
    k_add_ln<<<BB * GC, 256, 0, stream>>>(gh, ghtmp, p.ngg, p.ngb, gh);
    k_token_reg<<<BB, 256, 0, stream>>>(attn, reg);

    // a2: nodes attend to global tokens
    gemm(stream, xn, lw[l].a2in,                 p.a2_in_b,          vb,  NN,      DD, DD, 0);
    gemm(stream, gh, lw[l].a2in + 256 * 256,     p.a2_in_b + DD,     kh2, BB * GC, DD, DD, 0);
    gemm(stream, gh, lw[l].a2in + 2 * 256 * 256, p.a2_in_b + 2 * DD, vh2, BB * GC, DD, DD, 0);
    k_mha_n2g<<<NN * HH / 256, 256, 0, stream>>>(vb, kh2, vh2, skipb);
    gemm(stream, skipb, lw[l].a2out, p.a2_out_b, aggb, NN, DD, DD, 0);
    k_add_ln<<<NN, 256, 0, stream>>>(xn, aggb, p.nngg, p.nngb, xn);

    // ffn (relu fused into first WMMA gemm)
    gemm(stream, xn, lw[l].w1, p.b1, cat2, NN, DD, 2 * DD, 1);
    gemm(stream, cat2, lw[l].w2, p.b2, maskf, NN, 2 * DD, DD, 0);
    k_add_ln<<<NN, 256, 0, stream>>>(xn, maskf, p.n2g, p.n2b, xn);
  }

  // -------- outputs --------
  size_t total = ND + (size_t)BB * GC * DD + 1;
  k_write_out<<<(unsigned)((total + 255) / 256), 256, 0, stream>>>(xn, gh, reg,
                                                                   (float*)d_out, total);
}
